// GCN_6047313953618
// MI455X (gfx1250) — compile-verified, matrix operand order checked by reading, not run on hardware
//
#include <hip/hip_runtime.h>

#define NNODES 100000
#define NEDGES 1600000
#define IN_DIM 512
#define HID_DIM 128
#define OUT_DIM 64

typedef __attribute__((ext_vector_type(16))) __bf16 v16bf;
typedef __attribute__((ext_vector_type(8)))  float v8f;

union ABits { uint4 q[2]; v16bf v; };

// Pack two fp32 -> packed bf16 pair {hi,lo} with round-to-nearest (ties-away):
// one v_perm_b32 selecting the high halves after a +0x8000 rounding bias.
__device__ __forceinline__ unsigned int pack2bf(float lo, float hi) {
  unsigned int ul = __builtin_bit_cast(unsigned int, lo) + 0x8000u;
  unsigned int uh = __builtin_bit_cast(unsigned int, hi) + 0x8000u;
  return __builtin_amdgcn_perm(uh, ul, 0x07060302u);
}

// ---------------------------------------------------------------------------
// Repack fp32 W[K][N] (row-major) into the WMMA bf16 B-matrix lane layout:
// dword index i = (((nt*(K/32)+kt)*32 + lane)*8 + v)
// B VGPR v holds K = {koff, koff+1}, koff = (v<4 ? 2v : 16+2(v-4)) + 8*(lane>>4)
// ---------------------------------------------------------------------------
__global__ void pack_w_kernel(const float* __restrict__ W, unsigned int* __restrict__ Wp,
                              int K, int N) {
  int i = blockIdx.x * blockDim.x + threadIdx.x;
  int total = (N / 16) * (K / 32) * 32 * 8;
  if (i >= total) return;
  int v    = i & 7;
  int lane = (i >> 3) & 31;
  int rest = i >> 8;
  int kt   = rest % (K / 32);
  int nt   = rest / (K / 32);
  int h = lane >> 4, n = lane & 15;
  int koff = (v < 4 ? 2 * v : 16 + 2 * (v - 4)) + 8 * h;
  int kk  = kt * 32 + koff;
  int col = nt * 16 + n;
  Wp[i] = pack2bf(W[kk * N + col], W[(kk + 1) * N + col]);
}

// ---------------------------------------------------------------------------
// WMMA GEMM: C[M,N] = A[M,K] * B[K,N].
// Block = 256 threads = 8 waves, each wave owns one 16-row tile (M blocking).
// The 32xN bf16 B k-slice is staged in LDS once per block (double-buffered,
// conflict-free [buf][g][nt][lane][4] layout), with register prefetch of the
// next slice overlapping the WMMAs.
// ---------------------------------------------------------------------------
template <int K, int NT, bool A_IS_F32>
__global__ void gemm_wmma_kernel(const void* __restrict__ Aptr,
                                 const unsigned int* __restrict__ Bp,
                                 float* __restrict__ C, int Mtiles) {
  constexpr int KT = K / 32;
  __shared__ unsigned int bsh[2][2][NT][32][4];   // 2 buffers, g = (v<4 | v>=4)

  const int tid  = threadIdx.x;
  const int lane = tid & 31;
  const int wave = tid >> 5;
  const int tile = blockIdx.x * 8 + wave;
  const bool active = (tile < Mtiles);            // wave-uniform
  const int h = lane >> 4;
  const int m = lane & 15;
  const int N = NT * 16;
  const long row0 = (long)tile * 16;

  // cooperative B loader: thread t stages (nt = t>>5, lane) -> 8 dwords/slice
  const int bnt = tid >> 5;
  const bool bload = (bnt < NT);
  const unsigned int* bsrc = Bp + (bnt * KT) * 256 + lane * 8;

  uint4 s0 = {}, s1 = {};
  if (bload) {
    s0 = *(const uint4*)(bsrc + 0);
    s1 = *(const uint4*)(bsrc + 4);
    *(uint4*)(&bsh[0][0][bnt][lane][0]) = s0;
    *(uint4*)(&bsh[0][1][bnt][lane][0]) = s1;
  }
  __syncthreads();

  v8f acc[NT];
#pragma unroll
  for (int t = 0; t < NT; ++t) acc[t] = (v8f){};

#pragma unroll 2
  for (int kt = 0; kt < KT; ++kt) {
    const int cur = kt & 1, nxt = cur ^ 1;

    // prefetch next B slice into registers (overlaps WMMAs below)
    if (bload && (kt + 1 < KT)) {
      const unsigned int* p = bsrc + (kt + 1) * 256;
      s0 = *(const uint4*)(p + 0);
      s1 = *(const uint4*)(p + 4);
    }

    if (active) {
      // A tile: 16x32 bf16, VGPR v: K = (v<4 ? 2v : 16+2(v-4)) + 8h (+1)
      ABits a;
      if constexpr (A_IS_F32) {
        const float* arow = (const float*)Aptr + (row0 + m) * (long)K + kt * 32 + 8 * h;
        float4 f0 = *(const float4*)(arow + 0);
        float4 f1 = *(const float4*)(arow + 4);
        float4 f2 = *(const float4*)(arow + 16);
        float4 f3 = *(const float4*)(arow + 20);
        a.q[0] = make_uint4(pack2bf(f0.x, f0.y), pack2bf(f0.z, f0.w),
                            pack2bf(f1.x, f1.y), pack2bf(f1.z, f1.w));
        a.q[1] = make_uint4(pack2bf(f2.x, f2.y), pack2bf(f2.z, f2.w),
                            pack2bf(f3.x, f3.y), pack2bf(f3.z, f3.w));
      } else {
        const unsigned int* arow =
            (const unsigned int*)Aptr + (row0 + m) * (long)(K / 2) + kt * 16 + 4 * h;
        a.q[0] = *(const uint4*)(arow);
        a.q[1] = *(const uint4*)(arow + 8);
      }
#pragma unroll
      for (int nt = 0; nt < NT; ++nt) {
        ABits b;
        b.q[0] = *(const uint4*)(&bsh[cur][0][nt][lane][0]);
        b.q[1] = *(const uint4*)(&bsh[cur][1][nt][lane][0]);
        acc[nt] = __builtin_amdgcn_wmma_f32_16x16x32_bf16(
            /*neg_a=*/false, a.v, /*neg_b=*/false, b.v,
            /*c_mod=*/(short)0, acc[nt], /*reuse_a=*/false, /*reuse_b=*/false);
      }
    }

    if (bload && (kt + 1 < KT)) {
      *(uint4*)(&bsh[nxt][0][bnt][lane][0]) = s0;
      *(uint4*)(&bsh[nxt][1][bnt][lane][0]) = s1;
    }
    __syncthreads();
  }

  // C/D layout: VGPR v -> row m = v + 8h, col n = lane & 15
  if (active) {
#pragma unroll
    for (int nt = 0; nt < NT; ++nt)
#pragma unroll
      for (int v = 0; v < 8; ++v)
        C[(row0 + v + 8 * h) * (long)N + nt * 16 + (lane & 15)] = acc[nt][v];
  }
}

// ---------------------------------------------------------------------------
// buf[i] = bias[i & (D-1)]  (bias pre-load so scatter only adds messages)
// ---------------------------------------------------------------------------
template <int D>
__global__ void init_bias_kernel(float* __restrict__ buf, const float* __restrict__ b, long n) {
  long i = blockIdx.x * (long)blockDim.x + threadIdx.x;
  if (i >= n) return;
  buf[i] = b[i & (D - 1)];
}

// ---------------------------------------------------------------------------
// Edge scatter: agg[rows[e]] += vals[e] * support[cols[e]], 4 dims per thread.
// support/agg are L2-resident (51 MB / 192 MB L2) -> fp32 hw atomics at L2.
// ---------------------------------------------------------------------------
template <int D>
__global__ void scatter_edges_kernel(const int* __restrict__ rows, const int* __restrict__ cols,
                                     const float* __restrict__ vals,
                                     const float* __restrict__ support,
                                     float* __restrict__ agg, long nE) {
  constexpr int C = D / 4;
  constexpr int LOG2C = (D == 128) ? 5 : 4;
  long tid = blockIdx.x * (long)blockDim.x + threadIdx.x;
  long e = tid >> LOG2C;
  if (e >= nE) return;
  int c = (int)(tid & (C - 1));
  int r = rows[e], cl = cols[e];
  float v = vals[e];
  float4 s = *(const float4*)(support + (long)cl * D + c * 4);
  float* dst = agg + (long)r * D + c * 4;
  unsafeAtomicAdd(dst + 0, v * s.x);
  unsafeAtomicAdd(dst + 1, v * s.y);
  unsafeAtomicAdd(dst + 2, v * s.z);
  unsafeAtomicAdd(dst + 3, v * s.w);
}

// ---------------------------------------------------------------------------
// ReLU + convert to packed bf16 pairs (input to GEMM1's A path)
// ---------------------------------------------------------------------------
__global__ void relu_to_bf16_kernel(const float* __restrict__ a, unsigned int* __restrict__ hb,
                                    long npairs) {
  long i = blockIdx.x * (long)blockDim.x + threadIdx.x;
  if (i >= npairs) return;
  float x0 = fmaxf(a[2 * i + 0], 0.f);
  float x1 = fmaxf(a[2 * i + 1], 0.f);
  hb[i] = pack2bf(x0, x1);
}

extern "C" void kernel_launch(void* const* d_in, const int* in_sizes, int n_in,
                              void* d_out, int out_size, void* d_ws, size_t ws_size,
                              hipStream_t stream) {
  const float* x    = (const float*)d_in[0];
  const int*   rows = (const int*)d_in[1];
  const int*   cols = (const int*)d_in[2];
  const float* vals = (const float*)d_in[3];
  const float* W0   = (const float*)d_in[4];
  const float* b0   = (const float*)d_in[5];
  const float* W1   = (const float*)d_in[6];
  const float* b1   = (const float*)d_in[7];
  float* out = (float*)d_out;

  // Workspace layout (~128.2 MB total):
  char* ws = (char*)d_ws;
  float*        support = (float*)(ws);                        // 51.2 MB (reused for layer1)
  float*        agg0    = (float*)(ws + 51200000);             // 51.2 MB
  unsigned int* hb      = (unsigned int*)(ws + 102400000);     // 25.6 MB bf16 pairs
  unsigned int* w0p     = (unsigned int*)(ws + 128000000);     // 128 KB
  unsigned int* w1p     = (unsigned int*)(ws + 128131072);     // 16 KB

  const int Mtiles = NNODES / 16;          // 6250
  const int gemmBlocks = (Mtiles + 7) / 8; // 8 waves per block

  // 1) repack weights into WMMA bf16 B layout
  pack_w_kernel<<<(IN_DIM * HID_DIM / 2 + 255) / 256, 256, 0, stream>>>(W0, w0p, IN_DIM, HID_DIM);
  pack_w_kernel<<<(HID_DIM * OUT_DIM / 2 + 255) / 256, 256, 0, stream>>>(W1, w1p, HID_DIM, OUT_DIM);

  // 2) support = x @ W0   (fp32 A packed to bf16 in-register via v_perm)
  gemm_wmma_kernel<IN_DIM, HID_DIM / 16, true>
      <<<gemmBlocks, 256, 0, stream>>>(x, w0p, support, Mtiles);

  // 3) agg0 = b0 broadcast; scatter edges; ReLU -> bf16
  {
    long n = (long)NNODES * HID_DIM;
    init_bias_kernel<HID_DIM><<<(unsigned)((n + 255) / 256), 256, 0, stream>>>(agg0, b0, n);
    long t = (long)NEDGES * (HID_DIM / 4);
    scatter_edges_kernel<HID_DIM>
        <<<(unsigned)((t + 255) / 256), 256, 0, stream>>>(rows, cols, vals, support, agg0, NEDGES);
    long p = (long)NNODES * (HID_DIM / 2);
    relu_to_bf16_kernel<<<(unsigned)((p + 255) / 256), 256, 0, stream>>>(agg0, hb, p);
  }

  // 4) support1 = h @ W1 (bf16 A path; reuses `support` buffer)
  gemm_wmma_kernel<HID_DIM, OUT_DIM / 16, false>
      <<<gemmBlocks, 256, 0, stream>>>(hb, w1p, support, Mtiles);

  // 5) out = b1 broadcast; scatter edges into d_out
  {
    long n = (long)NNODES * OUT_DIM;
    init_bias_kernel<OUT_DIM><<<(unsigned)((n + 255) / 256), 256, 0, stream>>>(out, b1, n);
    long t = (long)NEDGES * (OUT_DIM / 4);
    scatter_edges_kernel<OUT_DIM>
        <<<(unsigned)((t + 255) / 256), 256, 0, stream>>>(rows, cols, vals, support, out, NEDGES);
  }
}
